// WRNN_91233695302481
// MI455X (gfx1250) — compile-verified, matrix operand order checked by reading, not run on hardware
//
#include <hip/hip_runtime.h>
#include <cstdint>
#include <cstddef>

// Problem constants (match reference)
#define TT 512
#define BB 128
#define II 512
#define HH 1024

typedef __attribute__((ext_vector_type(16))) __bf16 v16bf;
typedef __attribute__((ext_vector_type(8)))  float  v8f;
typedef int vsi4 __attribute__((vector_size(16)));   // matches builtin param type

#define WMMA_BF16(a, b, c) \
  __builtin_amdgcn_wmma_f32_16x16x32_bf16(false, (a), false, (b), (short)0, (c), false, false)

// Async global->LDS staging (gfx1250): guarded so a missing builtin degrades
// to a plain copy loop instead of breaking the build.
#if defined(__has_builtin)
#if __has_builtin(__builtin_amdgcn_global_load_async_to_lds_b128) && \
    __has_builtin(__builtin_amdgcn_s_wait_asynccnt)
#define HAVE_ASYNC_LDS 1
#endif
#endif

// Exact-typed pointer materialization (integer -> pointer casts are legal into
// any address space / pointee type).
#define GP4(p) ((__attribute__((address_space(1))) vsi4*)(uintptr_t)(p))
#define LP4(p) ((__attribute__((address_space(3))) vsi4*)(unsigned)(uintptr_t)(p))

static __device__ __forceinline__ __bf16 f2bf(float f) {
  unsigned u = __builtin_bit_cast(unsigned, f);
  unsigned r = u + 0x7FFFu + ((u >> 16) & 1u);   // round-to-nearest-even
  unsigned short h = (unsigned short)(r >> 16);
  return __builtin_bit_cast(__bf16, h);
}

// exp2-based tanh: 2 TRANS ops, clamped so exp2 cannot overflow.
static __device__ __forceinline__ float fast_tanh(float x) {
  x = fminf(fmaxf(x, -9.0f), 9.0f);
  float e = __builtin_amdgcn_exp2f(x * 2.8853900817779268f);   // exp(2x)
  return (e - 1.0f) * __builtin_amdgcn_rcpf(e + 1.0f);
}

// ---------------------------------------------------------------------------
// Fragment-swizzled layouts (per CDNA5 ISA 16-bit WMMA VGPR layouts, wave32):
// A (MxK): fragment (rowtile, ktile) = 512 contiguous bf16:
//   off = ((rowtile*(K/32) + ktile)*32 + lane)*16 + j
//   lane = kh*16 + (row&15), kh = (k>>3)&1, j = (k&7) | ((k&16)>>1)
// B (KxN): mirrored with column in place of row.
// C/pre (MxN fp32): fragment (rowtile, coltile) = 256 contiguous floats:
//   off = ((rowtile*(N/16) + coltile)*32 + lane)*8 + v   (lane/v = HW C layout)
// => every per-lane WMMA / epilogue operand is a contiguous 32B chunk.
// ---------------------------------------------------------------------------

__global__ __launch_bounds__(256) void swzB_kernel(
    const float* __restrict__ s1, const float* __restrict__ s2,
    __bf16* __restrict__ dst, int K, int N) {
  size_t d = (size_t)blockIdx.x * blockDim.x + threadIdx.x;
  if (d >= (size_t)K * N) return;
  int j    = (int)(d & 15);
  int lane = (int)((d >> 4) & 31);
  int kt   = (int)((d >> 9) % (size_t)(K >> 5));
  int ct   = (int)(d / ((size_t)(K >> 5) << 9));
  int kh = lane >> 4, cc = lane & 15;
  int kk = (j & 7) | ((j & 8) << 1) | (kh << 3);
  size_t src = (size_t)((kt << 5) + kk) * N + (ct << 4) + cc;
  float v = s1[src];
  if (s2) v += s2[src];
  dst[d] = f2bf(v);
}

__global__ __launch_bounds__(256) void swzA_kernel(
    const float* __restrict__ s1, const float* __restrict__ s2,
    __bf16* __restrict__ dst, int M, int K) {
  size_t d = (size_t)blockIdx.x * blockDim.x + threadIdx.x;
  if (d >= (size_t)M * K) return;
  int j    = (int)(d & 15);
  int lane = (int)((d >> 4) & 31);
  int kt   = (int)((d >> 9) % (size_t)(K >> 5));
  int rt   = (int)(d / ((size_t)(K >> 5) << 9));
  int kh = lane >> 4, r = lane & 15;
  int kk = (j & 7) | ((j & 8) << 1) | (kh << 3);
  size_t src = (size_t)(rt * 16 + r) * K + (kt << 5) + kk;
  float v = s1[src];
  if (s2) v = (v + s2[src]) * 0.5f;   // 2-tap moving average path for sx1
  dst[d] = f2bf(v);
}

// ---------------------------------------------------------------------------
// bf16 WMMA GEMM: C_frag[M,N] (fp32 fragment layout) = A_swz @ B_swz (+C).
// Block = 256 threads = 8 waves as 4(M) x 2(N); block tile = 128 x 128.
// Each wave: 2 rowtiles x 4 coltiles = 8 WMMA per K-step, 6 fragment loads.
// KT is compile-time so fragment addresses become immediate offsets.
// ---------------------------------------------------------------------------
template <int KT>
__global__ __launch_bounds__(256) void gemm_wmma_bf16(
    const __bf16* __restrict__ A, const __bf16* __restrict__ Bw,
    float* __restrict__ Cf, int Ntiles, int accumulate) {
  const int lane = threadIdx.x & 31;
  const int wave = threadIdx.x >> 5;
  const int wm = wave & 3, wn = wave >> 2;

  const int rt0 = blockIdx.x * 8 + wm * 2;        // two rowtiles: rt0, rt0+1
  const int ct0 = blockIdx.y * 8 + wn * 4;        // four coltiles: ct0..ct0+3

  const v8f vz = {0.f, 0.f, 0.f, 0.f, 0.f, 0.f, 0.f, 0.f};
  v8f acc[2][4] = {{vz, vz, vz, vz}, {vz, vz, vz, vz}};

  const v16bf* __restrict__ A0 = (const v16bf*)A + (size_t)rt0 * KT * 32 + lane;
  const v16bf* __restrict__ Bb = (const v16bf*)Bw + (size_t)ct0 * KT * 32 + lane;

#pragma unroll 4
  for (int kt = 0; kt < KT; ++kt) {
    v16bf a0 = A0[kt * 32];
    v16bf a1 = A0[(KT + kt) * 32];
    v16bf b0 = Bb[(0 * KT + kt) * 32];
    v16bf b1 = Bb[(1 * KT + kt) * 32];
    v16bf b2 = Bb[(2 * KT + kt) * 32];
    v16bf b3 = Bb[(3 * KT + kt) * 32];
    acc[0][0] = WMMA_BF16(a0, b0, acc[0][0]);
    acc[1][0] = WMMA_BF16(a1, b0, acc[1][0]);
    acc[0][1] = WMMA_BF16(a0, b1, acc[0][1]);
    acc[1][1] = WMMA_BF16(a1, b1, acc[1][1]);
    acc[0][2] = WMMA_BF16(a0, b2, acc[0][2]);
    acc[1][2] = WMMA_BF16(a1, b2, acc[1][2]);
    acc[0][3] = WMMA_BF16(a0, b3, acc[0][3]);
    acc[1][3] = WMMA_BF16(a1, b3, acc[1][3]);
  }

#pragma unroll
  for (int i = 0; i < 2; ++i) {
#pragma unroll
    for (int jx = 0; jx < 4; ++jx) {
      size_t base = (((size_t)(rt0 + i) * Ntiles + (ct0 + jx)) * 32 + lane) * 8;
      v8f* cp = (v8f*)(Cf + base);
      if (accumulate) { v8f old = *cp; *cp = old + acc[i][jx]; }
      else            { *cp = acc[i][jx]; }
    }
  }
}

// ---------------------------------------------------------------------------
// Persistent recurrence: h_t = tanh(pre[t] + h_{t-1} @ Wh), Tsteps serial.
// Grid = 32 blocks (one 32-column strip of H each), 256 threads (8 waves).
// Wh strip (64KB) staged ONCE into LDS via async global->LDS (ASYNCcnt);
// h bf16 A-swizzled in L2; device-wide barrier per step; pre read in
// fragment layout (v8f loads) and prefetched before the K-loop.
// ---------------------------------------------------------------------------
__global__ __launch_bounds__(256) void wrnn_rec(
    const float* __restrict__ pre, const __bf16* __restrict__ Wswz,
    float* __restrict__ outf, __bf16* __restrict__ hswz,
    const __bf16* __restrict__ zbuf, unsigned* bar,
    int Tsteps, int persist, int nblocks) {
  __shared__ __attribute__((aligned(32))) __bf16 Blds[2 * 32 * 512];  // 64 KB

  const int lane = threadIdx.x & 31;
  const int wave = threadIdx.x >> 5;
  const int bx = blockIdx.x;        // column strip: cols [bx*32, bx*32+32)
  const int KT = HH / 32;           // 32 K-tiles
  const int NT = HH / 16;           // 64 coltiles in fragment layout

  // Stage loop-invariant Wh strip into LDS (contiguous in swizzled layout).
  {
    const char* gsrc = (const char*)((const v16bf*)Wswz + (size_t)(2 * bx) * KT * 32);
    char* ldst = (char*)Blds;
#if defined(HAVE_ASYNC_LDS)
    for (int i = threadIdx.x; i < 4096; i += blockDim.x)   // 4096 x 16B = 64KB
      __builtin_amdgcn_global_load_async_to_lds_b128(
          GP4(gsrc + (size_t)i * 16), LP4(ldst + (size_t)i * 16), 0, 0);
    __builtin_amdgcn_s_wait_asynccnt(0);
#else
    const v16bf* src = (const v16bf*)gsrc;
    v16bf* dst = (v16bf*)ldst;
    for (int i = threadIdx.x; i < 2 * KT * 32; i += blockDim.x) dst[i] = src[i];
#endif
  }
  __syncthreads();

  const v8f vz = {0.f, 0.f, 0.f, 0.f, 0.f, 0.f, 0.f, 0.f};
  const size_t stepV = (size_t)8 * KT * 32;          // v16bf units per timestep slab
  const v16bf* hbaseV = (const v16bf*)hswz;
  const v16bf* zV = (const v16bf*)zbuf;
  volatile unsigned* gen = bar + 16;

  for (int t = 0; t < Tsteps; ++t) {
    // pre fragments this block/wave needs at the end of the step.
    size_t pb0 = (((size_t)(t * 8 + wave) * NT + (bx * 2 + 0)) * 32 + lane) * 8;
    size_t pb1 = (((size_t)(t * 8 + wave) * NT + (bx * 2 + 1)) * 32 + lane) * 8;
    __builtin_prefetch(pre + pb0, 0, 1);   // global_prefetch_b8
    __builtin_prefetch(pre + pb1, 0, 1);

    const v16bf* hr = (t == 0)
        ? zV
        : hbaseV + (size_t)(persist ? (t - 1) : ((t - 1) & 1)) * stepV;

    v8f acc0 = vz, acc1 = vz;
    const v16bf* Ab = hr + (size_t)wave * KT * 32 + lane;
    const v16bf* B0 = (const v16bf*)Blds + lane;
    const v16bf* B1 = (const v16bf*)Blds + (size_t)KT * 32 + lane;
#pragma unroll 4
    for (int kt = 0; kt < KT; ++kt) {
      v16bf a  = Ab[kt * 32];
      v16bf b0 = B0[kt * 32];
      v16bf b1 = B1[kt * 32];
      acc0 = WMMA_BF16(a, b0, acc0);
      acc1 = WMMA_BF16(a, b1, acc1);
    }

    // Fused epilogue: tanh(pre + acc) -> fp32 out (row-major) + bf16 h_t (A-swz).
    __bf16* hw = hswz + (size_t)(persist ? t : (t & 1)) * (stepV * 16);
    const int mloc = wave * 16 + (lane >> 4) * 8;
    v8f p0 = *(const v8f*)(pre + pb0);
    v8f p1 = *(const v8f*)(pre + pb1);
#pragma unroll
    for (int c = 0; c < 2; ++c) {
      int n  = bx * 32 + c * 16 + (lane & 15);
      int kk = n & 31;
      int kh = (kk >> 3) & 1;
      int j  = (kk & 7) | ((kk & 16) >> 1);
      const v8f& a = c ? acc1 : acc0;
      const v8f& p = c ? p1 : p0;
#pragma unroll
      for (int v = 0; v < 8; ++v) {
        int row = mloc + v;                           // 0..127
        float val = fast_tanh(p[v] + a[v]);
        outf[((size_t)t * BB + row) * HH + n] = val;
        int r = row & 15;
        size_t off = (((size_t)wave * KT + bx) * 32 + (kh * 16 + r)) * 16 + j;
        hw[off] = f2bf(val);
      }
    }

    // Device-wide barrier: all blocks finish step t before any reads h_t.
    __syncthreads();
    if (threadIdx.x == 0) {
      __threadfence();
      unsigned g0 = *gen;
      if (atomicAdd(bar, 1u) == (unsigned)nblocks - 1u) {
        *bar = 0u;
        __threadfence();
        atomicAdd((unsigned*)(bar + 16), 1u);
      } else {
        while (*gen == g0) __builtin_amdgcn_s_sleep(4);
      }
    }
    __syncthreads();
  }
}

// ---------------------------------------------------------------------------
extern "C" void kernel_launch(void* const* d_in, const int* in_sizes, int n_in,
                              void* d_out, int out_size, void* d_ws, size_t ws_size,
                              hipStream_t stream) {
  (void)in_sizes; (void)n_in; (void)out_size; (void)ws_size;
  const float* x   = (const float*)d_in[0];
  const float* Wi0 = (const float*)d_in[1];
  const float* Wh0 = (const float*)d_in[2];
  const float* Ws0 = (const float*)d_in[3];
  const float* Wi1 = (const float*)d_in[4];
  const float* Wh1 = (const float*)d_in[5];
  const float* Ws1 = (const float*)d_in[6];

  float* out0 = (float*)d_out;                     // [T,B,H]
  float* out1 = out0 + (size_t)TT * BB * HH;       // [T-1,B,H]

  // Carve workspace (256B aligned)
  uint8_t* p = (uint8_t*)d_ws;
  auto carve = [&](size_t bytes) -> void* {
    void* r = p; p += (bytes + 255) & ~(size_t)255; return r;
  };
  __bf16*  xbA   = (__bf16*)carve((size_t)TT * BB * II * 2);
  __bf16*  sxA   = (__bf16*)carve((size_t)(TT - 1) * BB * II * 2);
  __bf16*  W0cB  = (__bf16*)carve((size_t)II * HH * 2);
  __bf16*  Wh0B  = (__bf16*)carve((size_t)HH * HH * 2);
  __bf16*  Wi1B  = (__bf16*)carve((size_t)HH * HH * 2);
  __bf16*  Wh1B  = (__bf16*)carve((size_t)HH * HH * 2);
  __bf16*  Ws1B  = (__bf16*)carve((size_t)II * HH * 2);
  float*   pre   = (float*)carve((size_t)TT * BB * HH * 4);   // fragment layout, reused
  __bf16*  o0b   = (__bf16*)carve((size_t)TT * BB * HH * 2);  // bf16 out0, A-swizzled
  __bf16*  hping = (__bf16*)carve((size_t)2 * BB * HH * 2);
  __bf16*  zbuf  = (__bf16*)carve((size_t)BB * HH * 2);
  unsigned* bar  = (unsigned*)carve(256);

  // --- Weight / input conversion into fragment-swizzled bf16 ---
  {
    size_t n = (size_t)II * HH;
    swzB_kernel<<<(unsigned)(n / 256), 256, 0, stream>>>(Wi0, Ws0, W0cB, II, HH);
    swzB_kernel<<<(unsigned)(n / 256), 256, 0, stream>>>(Ws1, nullptr, Ws1B, II, HH);
  }
  {
    size_t n = (size_t)HH * HH;
    swzB_kernel<<<(unsigned)(n / 256), 256, 0, stream>>>(Wh0, nullptr, Wh0B, HH, HH);
    swzB_kernel<<<(unsigned)(n / 256), 256, 0, stream>>>(Wi1, nullptr, Wi1B, HH, HH);
    swzB_kernel<<<(unsigned)(n / 256), 256, 0, stream>>>(Wh1, nullptr, Wh1B, HH, HH);
  }
  {
    size_t n = (size_t)TT * BB * II;
    swzA_kernel<<<(unsigned)(n / 256), 256, 0, stream>>>(x, nullptr, xbA, TT * BB, II);
  }
  {
    size_t n = (size_t)(TT - 1) * BB * II;
    swzA_kernel<<<(unsigned)(n / 256), 256, 0, stream>>>(
        x, x + (size_t)BB * II, sxA, (TT - 1) * BB, II);
  }
  (void)hipMemsetAsync(zbuf, 0, (size_t)BB * HH * 2, stream);

  // --- Layer 0: pre0 = x @ (Wi0+Ws0), then recurrence ---
  gemm_wmma_bf16<II / 32><<<dim3((TT * BB) / 128, HH / 128), 256, 0, stream>>>(
      xbA, W0cB, pre, HH / 16, 0);
  (void)hipMemsetAsync(bar, 0, 256, stream);
  wrnn_rec<<<HH / 32, 256, 0, stream>>>(pre, Wh0B, out0, o0b, zbuf, bar,
                                        TT, /*persist=*/1, HH / 32);

  // --- Layer 1: pre1 = out0[1:] @ Wi1 + sx1 @ Ws1, then recurrence ---
  // out0[1:] in A-swizzled layout = skip 8 rowtiles (128 rows) = 8*32*512 elems.
  gemm_wmma_bf16<HH / 32><<<dim3(((TT - 1) * BB) / 128, HH / 128), 256, 0, stream>>>(
      o0b + (size_t)8 * (HH / 32) * 512, Wi1B, pre, HH / 16, 0);
  gemm_wmma_bf16<II / 32><<<dim3(((TT - 1) * BB) / 128, HH / 128), 256, 0, stream>>>(
      sxA, Ws1B, pre, HH / 16, 1);
  (void)hipMemsetAsync(bar, 0, 256, stream);
  wrnn_rec<<<HH / 32, 256, 0, stream>>>(pre, Wh1B, out1, hping, zbuf, bar,
                                        TT - 1, /*persist=*/0, HH / 32);
}
